// BiMambaBlock_73040213835812
// MI455X (gfx1250) — compile-verified, matrix operand order checked by reading, not run on hardware
//
#include <hip/hip_runtime.h>
#include <hip/hip_bf16.h>

// ---- problem constants (match reference) ----
#define D_MODEL 768
#define D_INNER 1536
#define N_STATE 16
#define DT_RANK 96
#define K_CONV  4
#define BATCH   2
#define LSEQ    1024
#define ROWS    (BATCH * LSEQ)           // 2048 token rows
#define XPROJ_N (DT_RANK + 2 * N_STATE)  // 128

// CDNA5 async global->LDS copies (inline asm; assembled OK in round 2)
#define ASYNC_LDS_COPY 1

typedef __attribute__((ext_vector_type(16))) _Float16 v16h;
typedef __attribute__((ext_vector_type(8)))  _Float16 v8h;
typedef __attribute__((ext_vector_type(8)))  float    v8f;

#define CAT16(a, b) __builtin_shufflevector(a, b, 0,1,2,3,4,5,6,7,8,9,10,11,12,13,14,15)

__device__ __forceinline__ float silu_f(float x) { return x / (1.0f + __expf(-x)); }
__device__ __forceinline__ float softplus_f(float x) {
  return (x > 20.0f) ? x : log1pf(__expf(x));
}

// 16-byte async copy: global -> LDS (ASYNCcnt-tracked DMA path on CDNA5)
__device__ __forceinline__ void async_copy16(const _Float16* g, _Float16* l) {
#if ASYNC_LDS_COPY
  unsigned lds = (unsigned)(unsigned long long)l;           // low 32 bits = LDS byte addr
  unsigned long long ga = (unsigned long long)g;
  asm volatile("global_load_async_to_lds_b128 %0, %1, off"
               :: "v"(lds), "v"(ga) : "memory");
#else
  *(v8h*)l = *(const v8h*)g;
#endif
}

__device__ __forceinline__ void wait_async_lds() {
#if ASYNC_LDS_COPY
  asm volatile("s_wait_asynccnt 0x0" ::: "memory");
#endif
}

// ---------------------------------------------------------------------------
// fp32 -> f16 conversion (weights, activations)
// ---------------------------------------------------------------------------
__global__ void cvt_f32_to_f16_kernel(const float* __restrict__ src,
                                      _Float16* __restrict__ dst, int n) {
  int i = blockIdx.x * blockDim.x + threadIdx.x;
  if (i < n) dst[i] = (_Float16)src[i];
}

// ---------------------------------------------------------------------------
// LayerNorm over D_MODEL, emits f16 (and a time-reversed f16 copy for bwd)
// ---------------------------------------------------------------------------
__global__ void layernorm_kernel(const float* __restrict__ u,
                                 const float* __restrict__ w,
                                 const float* __restrict__ b,
                                 _Float16* __restrict__ un_h,
                                 _Float16* __restrict__ un_rev_h) {
  __shared__ float red[256];
  const int row = blockIdx.x;            // 0..ROWS-1
  const int t   = threadIdx.x;           // 0..255
  const float* x = u + (size_t)row * D_MODEL;

  float s = 0.0f;
  for (int i = t; i < D_MODEL; i += 256) s += x[i];
  red[t] = s; __syncthreads();
  for (int off = 128; off > 0; off >>= 1) {
    if (t < off) red[t] += red[t + off];
    __syncthreads();
  }
  const float mu = red[0] * (1.0f / D_MODEL);
  __syncthreads();

  float v = 0.0f;
  for (int i = t; i < D_MODEL; i += 256) { float d = x[i] - mu; v += d * d; }
  red[t] = v; __syncthreads();
  for (int off = 128; off > 0; off >>= 1) {
    if (t < off) red[t] += red[t + off];
    __syncthreads();
  }
  const float rstd = rsqrtf(red[0] * (1.0f / D_MODEL) + 1e-5f);

  const int bb = row / LSEQ, l = row % LSEQ;
  const int rrow = bb * LSEQ + (LSEQ - 1 - l);
  for (int i = t; i < D_MODEL; i += 256) {
    float y = (x[i] - mu) * rstd * w[i] + b[i];
    _Float16 h = (_Float16)y;
    un_h[(size_t)row * D_MODEL + i]      = h;
    un_rev_h[(size_t)rrow * D_MODEL + i] = h;
  }
}

// ---------------------------------------------------------------------------
// LDS-staged, double-buffered WMMA GEMM:
//   C[M,N](f32) = A[M,K](f16,row) * W[N,K]^T(f16,row)   [+ resid]
//
// Block: 256 threads = 8 waves as 2(M) x 4(N). Macro-tile 64M x 128N.
// Per wave: 32x32 output = 2x2 WMMA tiles. Per K-step(32): async-DMA the
// NEXT A(64x32)+B(128x32) tiles into the other LDS buffer while computing
// 4x v_wmma_f32_16x16x32_f16 on the current one. One barrier per K-step.
//
// N, K compile-time -> constant address math, straight-line epilogue.
// ---------------------------------------------------------------------------
template <int N, int K, bool RESID>
__global__ void wmma_gemm_nt_kernel(const _Float16* __restrict__ A,
                                    const _Float16* __restrict__ W,
                                    float* __restrict__ C,
                                    const float* __restrict__ resid) {
  __shared__ _Float16 ldsA[2][64 * 32];     // 2 x 4 KB
  __shared__ _Float16 ldsB[2][128 * 32];    // 2 x 8 KB

  const int tid  = threadIdx.x;
  const int lane = tid & 31;
  const int w    = tid >> 5;            // 0..7
  const int wm   = w >> 2;              // 0..1  (M slot)
  const int wn   = w & 3;               // 0..3  (N slot)
  const int r    = lane & 15;
  const int hi   = lane >> 4;

  const int m0 = blockIdx.y * 64;
  const int n0 = blockIdx.x * 128;

  // per-thread staging assignment: one 16B chunk of A, two 16B chunks of B
  const int crow  = tid >> 2;           // 0..63
  const int cpart = (tid & 3) * 8;      // half-offset within row (0,8,16,24)

  const _Float16* gA  = A + (size_t)(m0 + crow) * K + cpart;
  const _Float16* gB0 = W + (size_t)(n0 + crow) * K + cpart;
  const _Float16* gB1 = W + (size_t)(n0 + 64 + crow) * K + cpart;

  auto stage = [&](int k0, int buf) {
    async_copy16(gA  + k0, &ldsA[buf][crow * 32 + cpart]);
    async_copy16(gB0 + k0, &ldsB[buf][crow * 32 + cpart]);
    async_copy16(gB1 + k0, &ldsB[buf][(64 + crow) * 32 + cpart]);
  };

  constexpr int STEPS = K / 32;
  v8f acc[2][2] = {};

  stage(0, 0);
  for (int s = 0; s < STEPS; ++s) {
    const int cur = s & 1;
    wait_async_lds();        // our async writes (into buf cur) done
    __syncthreads();         // all waves' writes done; prior reads of cur^1 done
    if (s + 1 < STEPS) stage((s + 1) * 32, cur ^ 1);  // prefetch next tile

    v16h afrag[2], bfrag[2];
#pragma unroll
    for (int t = 0; t < 2; ++t) {
      const _Float16* ap = &ldsA[cur][(wm * 32 + t * 16 + r) * 32 + hi * 8];
      v8h a0 = *(const v8h*)ap;
      v8h a1 = *(const v8h*)(ap + 16);
      afrag[t] = CAT16(a0, a1);
      const _Float16* bp = &ldsB[cur][(wn * 32 + t * 16 + r) * 32 + hi * 8];
      v8h b0 = *(const v8h*)bp;
      v8h b1 = *(const v8h*)(bp + 16);
      bfrag[t] = CAT16(b0, b1);
    }

#pragma unroll
    for (int tm = 0; tm < 2; ++tm)
#pragma unroll
      for (int tn = 0; tn < 2; ++tn)
        acc[tm][tn] = __builtin_amdgcn_wmma_f32_16x16x32_f16(
            false, afrag[tm], false, bfrag[tn],
            (short)0, acc[tm][tn], false, false);
  }

  // epilogue: C/D layout -> VGPR v at lane hi*16+r is row hi*8+v, col r
#pragma unroll
  for (int tm = 0; tm < 2; ++tm) {
    const int row0 = m0 + wm * 32 + tm * 16 + hi * 8;
#pragma unroll
    for (int tn = 0; tn < 2; ++tn) {
      const int col = n0 + wn * 32 + tn * 16 + r;
      float* cp = C + (size_t)row0 * N + col;
      const float* rp = RESID ? (resid + (size_t)row0 * N + col) : nullptr;
#pragma unroll
      for (int v = 0; v < 8; ++v) {
        float val = acc[tm][tn][v];
        if constexpr (RESID) val += rp[(size_t)v * N];
        cp[(size_t)v * N] = val;
      }
    }
  }
}

// ---------------------------------------------------------------------------
// depthwise causal conv (K_CONV taps) + SiLU
// ---------------------------------------------------------------------------
__global__ void conv_silu_kernel(const float* __restrict__ xz,
                                 const float* __restrict__ cw,
                                 const float* __restrict__ cb,
                                 float* __restrict__ xc,
                                 _Float16* __restrict__ xc_h) {
  const int idx = blockIdx.x * blockDim.x + threadIdx.x;
  if (idx >= ROWS * D_INNER) return;
  const int d   = idx % D_INNER;
  const int row = idx / D_INNER;
  const int b = row / LSEQ, l = row % LSEQ;

  float s = cb[d];
#pragma unroll
  for (int k = 0; k < K_CONV; ++k) {
    const int lp = l - (K_CONV - 1) + k;
    if (lp >= 0)
      s += xz[(size_t)(b * LSEQ + lp) * (2 * D_INNER) + d] * cw[d * K_CONV + k];
  }
  s = silu_f(s);
  xc[idx]   = s;
  xc_h[idx] = (_Float16)s;
}

// ---------------------------------------------------------------------------
// extract dt_raw (first DT_RANK cols of x_dbl) as f16 for the dt GEMM
// ---------------------------------------------------------------------------
__global__ void extract_dtraw_kernel(const float* __restrict__ x_dbl,
                                     _Float16* __restrict__ dtraw_h) {
  const int i = blockIdx.x * blockDim.x + threadIdx.x;
  if (i >= ROWS * DT_RANK) return;
  const int row = i / DT_RANK, c = i % DT_RANK;
  dtraw_h[i] = (_Float16)x_dbl[(size_t)row * XPROJ_N + c];
}

// ---------------------------------------------------------------------------
// selective scan: 16 lanes per (b,d), one SSM state each; lane-group shfl
// reduction for y = sum_n h_n * C_n.
// ---------------------------------------------------------------------------
__global__ void ssm_scan_kernel(const float* __restrict__ dt_pre,
                                const float* __restrict__ xc,
                                const float* __restrict__ x_dbl,
                                const float* __restrict__ xz,
                                const float* __restrict__ A_log,
                                const float* __restrict__ dt_bias,
                                const float* __restrict__ Dp,
                                _Float16* __restrict__ y_cat,
                                int col_off, int reverse) {
  const int tid = blockIdx.x * blockDim.x + threadIdx.x;
  if (tid >= BATCH * D_INNER * N_STATE) return;
  const int n  = tid & (N_STATE - 1);
  const int bd = tid >> 4;
  const int b = bd / D_INNER, d = bd % D_INNER;

  const float A   = -__expf(A_log[d * N_STATE + n]);
  const float dtb = dt_bias[d];
  const float Dd  = Dp[d];
  float h = 0.0f;

  for (int l = 0; l < LSEQ; ++l) {
    const size_t row = (size_t)b * LSEQ + l;
    const float dt = softplus_f(dt_pre[row * D_INNER + d] + dtb);
    const float xv = xc[row * D_INNER + d];
    const float* xd = x_dbl + row * XPROJ_N;

    h = __expf(dt * A) * h + (dt * xv) * xd[DT_RANK + n];
    float y = h * xd[DT_RANK + N_STATE + n];

    y += __shfl_xor(y, 1, 16);
    y += __shfl_xor(y, 2, 16);
    y += __shfl_xor(y, 4, 16);
    y += __shfl_xor(y, 8, 16);

    if (n == 0) {
      y += xv * Dd;
      const float zv = xz[row * (2 * D_INNER) + D_INNER + d];
      y *= silu_f(zv);
      const size_t orow = reverse ? ((size_t)b * LSEQ + (LSEQ - 1 - l)) : row;
      y_cat[orow * (2 * D_INNER) + col_off + d] = (_Float16)y;
    }
  }
}

// ---------------------------------------------------------------------------
// launcher
// ---------------------------------------------------------------------------
extern "C" void kernel_launch(void* const* d_in, const int* in_sizes, int n_in,
                              void* d_out, int out_size, void* d_ws, size_t ws_size,
                              hipStream_t stream) {
  (void)in_sizes; (void)n_in; (void)out_size; (void)ws_size;

  const float* u       = (const float*)d_in[0];
  const float* norm_w  = (const float*)d_in[1];
  const float* norm_b  = (const float*)d_in[2];
  const float* in_w[2]    = {(const float*)d_in[3],  (const float*)d_in[11]};
  const float* conv_w[2]  = {(const float*)d_in[4],  (const float*)d_in[12]};
  const float* conv_b[2]  = {(const float*)d_in[5],  (const float*)d_in[13]};
  const float* A_log[2]   = {(const float*)d_in[6],  (const float*)d_in[14]};
  const float* xproj_w[2] = {(const float*)d_in[7],  (const float*)d_in[15]};
  const float* dt_w[2]    = {(const float*)d_in[8],  (const float*)d_in[16]};
  const float* dt_b[2]    = {(const float*)d_in[9],  (const float*)d_in[17]};
  const float* Dp[2]      = {(const float*)d_in[10], (const float*)d_in[18]};
  const float* out_w      = (const float*)d_in[19];
  float* out = (float*)d_out;

  char* ws = (char*)d_ws;
  size_t off = 0;
  auto alloc = [&](size_t bytes) -> char* {
    char* p = ws + off;
    off += (bytes + 255) & ~(size_t)255;
    return p;
  };

  _Float16* un_h     = (_Float16*)alloc((size_t)ROWS * D_MODEL * 2);
  _Float16* un_rev_h = (_Float16*)alloc((size_t)ROWS * D_MODEL * 2);
  _Float16* in_w_h[2]    = {(_Float16*)alloc((size_t)2 * D_INNER * D_MODEL * 2),
                            (_Float16*)alloc((size_t)2 * D_INNER * D_MODEL * 2)};
  _Float16* xproj_w_h[2] = {(_Float16*)alloc((size_t)XPROJ_N * D_INNER * 2),
                            (_Float16*)alloc((size_t)XPROJ_N * D_INNER * 2)};
  _Float16* dt_w_h[2]    = {(_Float16*)alloc((size_t)D_INNER * DT_RANK * 2),
                            (_Float16*)alloc((size_t)D_INNER * DT_RANK * 2)};
  _Float16* out_w_h  = (_Float16*)alloc((size_t)D_MODEL * 2 * D_INNER * 2);
  float*    xz       = (float*)alloc((size_t)ROWS * 2 * D_INNER * 4);
  float*    xc       = (float*)alloc((size_t)ROWS * D_INNER * 4);
  _Float16* xc_h     = (_Float16*)alloc((size_t)ROWS * D_INNER * 2);
  float*    x_dbl    = (float*)alloc((size_t)ROWS * XPROJ_N * 4);
  _Float16* dtraw_h  = (_Float16*)alloc((size_t)ROWS * DT_RANK * 2);
  float*    dt_pre   = (float*)alloc((size_t)ROWS * D_INNER * 4);
  _Float16* y_cat_h  = (_Float16*)alloc((size_t)ROWS * 2 * D_INNER * 2);

  auto cvt = [&](const float* s, _Float16* d, int n) {
    cvt_f32_to_f16_kernel<<<(n + 255) / 256, 256, 0, stream>>>(s, d, n);
  };

  // --- weight conversions (f32 -> f16) ---
  for (int br = 0; br < 2; ++br) {
    cvt(in_w[br],    in_w_h[br],    2 * D_INNER * D_MODEL);
    cvt(xproj_w[br], xproj_w_h[br], XPROJ_N * D_INNER);
    cvt(dt_w[br],    dt_w_h[br],    D_INNER * DT_RANK);
  }
  cvt(out_w, out_w_h, D_MODEL * 2 * D_INNER);

  // --- layernorm (+ reversed copy) ---
  layernorm_kernel<<<ROWS, 256, 0, stream>>>(u, norm_w, norm_b, un_h, un_rev_h);

  const dim3 gemm_block(256);
  const int m_blocks = ROWS / 64;  // 32

  // --- the two Mamba branches (sequential, reusing scratch) ---
  for (int br = 0; br < 2; ++br) {
    const _Float16* un_sel = (br == 0) ? un_h : un_rev_h;

    // in-proj: [2048 x 768] x [3072 x 768]^T -> xz [2048 x 3072]
    wmma_gemm_nt_kernel<2 * D_INNER, D_MODEL, false>
        <<<dim3(2 * D_INNER / 128, m_blocks), gemm_block, 0, stream>>>(
            un_sel, in_w_h[br], xz, nullptr);

    // depthwise causal conv + SiLU
    conv_silu_kernel<<<(ROWS * D_INNER + 255) / 256, 256, 0, stream>>>(
        xz, conv_w[br], conv_b[br], xc, xc_h);

    // x-proj: [2048 x 1536] x [128 x 1536]^T -> x_dbl [2048 x 128]
    wmma_gemm_nt_kernel<XPROJ_N, D_INNER, false>
        <<<dim3(XPROJ_N / 128, m_blocks), gemm_block, 0, stream>>>(
            xc_h, xproj_w_h[br], x_dbl, nullptr);

    // dt_raw -> f16
    extract_dtraw_kernel<<<(ROWS * DT_RANK + 255) / 256, 256, 0, stream>>>(x_dbl, dtraw_h);

    // dt-proj: [2048 x 96] x [1536 x 96]^T -> dt_pre [2048 x 1536]
    wmma_gemm_nt_kernel<D_INNER, DT_RANK, false>
        <<<dim3(D_INNER / 128, m_blocks), gemm_block, 0, stream>>>(
            dtraw_h, dt_w_h[br], dt_pre, nullptr);

    // selective scan + gate, writes into its half of y_cat
    ssm_scan_kernel<<<(BATCH * D_INNER * N_STATE + 255) / 256, 256, 0, stream>>>(
        dt_pre, xc, x_dbl, xz, A_log[br], dt_b[br], Dp[br], y_cat_h,
        /*col_off=*/br * D_INNER, /*reverse=*/br);
  }

  // out-proj + residual: [2048 x 3072] x [768 x 3072]^T + u -> out [2048 x 768]
  wmma_gemm_nt_kernel<D_MODEL, 2 * D_INNER, true>
      <<<dim3(D_MODEL / 128, m_blocks), gemm_block, 0, stream>>>(
          y_cat_h, out_w_h, out, u);
}
